// ExperimentalGNN_4166118277634
// MI455X (gfx1250) — matrix operand order repaired and checked by reading, not compiled, not run on hardware
//
#include <hip/hip_runtime.h>
#include <hip/hip_bf16.h>

// ---------------- problem constants ----------------
constexpr int NN   = 10000;
constexpr int EEc  = 160000;
constexpr int GGc  = 64;
constexpr int DDc  = 512;
constexpr int HHc  = 8;
constexpr int CCc  = 64;
constexpr int LLc  = 8;
constexpr int DEc  = 256;            // DE = D/2
constexpr int EALL = EEc + NN;       // 170000 (= 10625*16)

typedef __attribute__((ext_vector_type(16))) __bf16 v16bf;
typedef __attribute__((ext_vector_type(8)))  __bf16 v8bf;
typedef __attribute__((ext_vector_type(4)))  __bf16 v4bf;
typedef __attribute__((ext_vector_type(8)))  float  v8f;
typedef __attribute__((ext_vector_type(4)))  unsigned v4u;
typedef __attribute__((ext_vector_type(8)))  int      v8i_t;
typedef __attribute__((ext_vector_type(4)))  int      v4i_t;

#define CDIV(a, b) (((a) + (b) - 1) / (b))

// Tensor Data Mover availability (probe-verified builtin; arity differs by toolchain)
#ifndef USE_TDM
#define USE_TDM 1
#endif
#if USE_TDM && defined(__has_builtin)
#if __has_builtin(__builtin_amdgcn_tensor_load_to_lds) && \
    __has_builtin(__builtin_amdgcn_s_wait_tensorcnt)
#define HAVE_TDM 1
#endif
#endif
#ifndef HAVE_TDM
#define HAVE_TDM 0
#endif

// ---------------- block-wide reduction (wave32 aware) ----------------
__device__ __forceinline__ float block_reduce_sum(float v, float* sbuf) {
#pragma unroll
  for (int off = 16; off > 0; off >>= 1) v += __shfl_xor(v, off, 32);
  const int lane = threadIdx.x & 31;
  const int wid  = threadIdx.x >> 5;
  const int nw   = (blockDim.x + 31) >> 5;
  if (lane == 0) sbuf[wid] = v;
  __syncthreads();
  float r = (threadIdx.x < (unsigned)nw) ? sbuf[threadIdx.x] : 0.f;
  if (threadIdx.x < 32) {
#pragma unroll
    for (int off = 16; off > 0; off >>= 1) r += __shfl_xor(r, off, 32);
    if (threadIdx.x == 0) sbuf[0] = r;
  }
  __syncthreads();
  r = sbuf[0];
  __syncthreads();
  return r;
}

// ---------------- fused Linear + LayerNorm + ELU (one block per row) ----------------
template <typename OutT>
__global__ void linear_ln_elu_kernel(const float* __restrict__ X, int xstride, int xoff,
                                     int kdim, const float* __restrict__ W,
                                     const float* __restrict__ b, OutT* __restrict__ Y) {
  __shared__ float sbuf[16];
  const int row   = blockIdx.x;
  const int j     = threadIdx.x;
  const int ncols = blockDim.x;
  const float* xr = X + (size_t)row * xstride + xoff;
  float v = b[j];
  for (int k = 0; k < kdim; ++k) v += xr[k] * W[(size_t)k * ncols + j];
  const float inv_n = 1.f / (float)ncols;
  float mean = block_reduce_sum(v, sbuf) * inv_n;
  float d    = v - mean;
  float var  = block_reduce_sum(d * d, sbuf) * inv_n;
  float z    = d * rsqrtf(var + 1e-5f);
  float e    = z > 0.f ? z : (expf(z) - 1.f);
  Y[(size_t)row * ncols + j] = (OutT)e;
}

// ---------------- conversion / transpose helpers ----------------
__global__ void cvt_bf16_kernel(const float* __restrict__ X, __bf16* __restrict__ Y, int n4) {
  int t = blockIdx.x * blockDim.x + threadIdx.x;
  if (t >= n4) return;
  float4 f = ((const float4*)X)[t];
  v4bf v;
  v[0] = (__bf16)f.x; v[1] = (__bf16)f.y; v[2] = (__bf16)f.z; v[3] = (__bf16)f.w;
  ((v4bf*)Y)[t] = v;
}

// WT[n][k] = (bf16)W[k][n];  W is K x Nd row-major.
__global__ void transpose_cvt_kernel(const float* __restrict__ W, __bf16* __restrict__ WT,
                                     int K, int Nd) {
  int t = blockIdx.x * blockDim.x + threadIdx.x;
  if (t >= K * Nd) return;
  int n = t / K, k = t - n * K;
  WT[(size_t)n * K + k] = (__bf16)W[(size_t)k * Nd + n];
}

// ---------------- Tensor Data Mover: 2D tile load (rows x 32 bf16) ----------------
// LDS padding: +4 dwords (16B) after every 16 dwords (64B) => 80B row pitch.
#if HAVE_TDM
__device__ __forceinline__ void tdm_load_2d(const __bf16* gaddr, unsigned lds_off,
                                            int tile_rows, int row_stride_elems) {
  const unsigned long long ga = (unsigned long long)(size_t)gaddr;
  v4u g0;
  g0[0] = 1u;                                    // count=1, user descriptor
  g0[1] = lds_off;                               // lds_addr (bytes)
  g0[2] = (unsigned)ga;                          // global_addr[31:0]
  g0[3] = (unsigned)((ga >> 32) & 0x01FFFFFFu)   // global_addr[56:32]
          | 0x80000000u;                         // type = 2 ("image")
  v8i_t g1;
  g1[0] = (int)(0x00010000u                      // data_size = 1 -> 2 bytes
                | 0x00100000u                    // pad_enable
                | (3u << 22)                     // pad_interval: 16 dwords (64B)
                | (3u << 25));                   // pad_amount:   4 dwords (16B)
  g1[1] = (int)(32u << 16);                      // tensor_dim0 = 32   (bits 79:48)
  g1[2] = (int)((unsigned)tile_rows << 16);      // tensor_dim1        (bits 111:80)
  g1[3] = (int)(32u << 16);                      // tile_dim0 = 32     (bits 127:112)
  g1[4] = tile_rows;                             // tile_dim1
  g1[5] = row_stride_elems;                      // tensor_dim0_stride (low 32)
  g1[6] = 0;
  g1[7] = 0;
  v4i_t gz = {0, 0, 0, 0};
#if __clang_major__ >= 23
  v8i_t gz8 = {0, 0, 0, 0, 0, 0, 0, 0};
  __builtin_amdgcn_tensor_load_to_lds(g0, g1, gz, gz, gz8, 0);
#else
  __builtin_amdgcn_tensor_load_to_lds(g0, g1, gz, gz, 0);
#endif
}
#endif

// ---------------- WMMA GEMM (double-buffered TDM pipeline) ----------------
// C(MxNd) = A(MxK) @ B(KxNd).  A: bf16 row-major.  BT: bf16, B transposed (Nd x K).
// 128 threads = 4 waves; block tile 16(M) x 64(N); wave w owns cols [16w, 16w+16).
// Ping-pong LDS tiles; wave0 DMAs next B tile, wave1 DMAs next A tile, then
// s_wait_tensorcnt(1): TENSORcnt completes in-order per wave, so the *current*
// tile's DMA is guaranteed done while the next stays in flight behind the WMMA.
constexpr int TM = 16, TN = 64, TK = 32;
constexpr int LP = 40;   // padded LDS row pitch in bf16 elements (80B)

template <int K, bool EDOT>
__global__ __launch_bounds__(128) void gemm_wmma_kernel(
    const __bf16* __restrict__ A, const __bf16* __restrict__ BT,
    float* __restrict__ C, const float* __restrict__ aE, float* __restrict__ s_e,
    int Nd) {
  __shared__ alignas(16) __bf16 sA[2][TM][LP];
  __shared__ alignas(16) __bf16 sBt[2][TN][LP];
  const int tid  = threadIdx.x;
  const int wave = tid >> 5, lane = tid & 31;
  const int half = lane >> 4, mn = lane & 15;
  const int row0 = blockIdx.x * TM;
  const int col0 = blockIdx.y * TN;
  constexpr int NT = K / TK;
  v8f acc = {0.f, 0.f, 0.f, 0.f, 0.f, 0.f, 0.f, 0.f};

#if HAVE_TDM
  // prologue: DMA tile 0 into buffer 0
  if (wave == 0)
    tdm_load_2d(&BT[(size_t)col0 * K], (unsigned)(size_t)&sBt[0][0][0], TN, K);
  else if (wave == 1)
    tdm_load_2d(&A[(size_t)row0 * K], (unsigned)(size_t)&sA[0][0][0], TM, K);
#else
  {  // manual stage of tile 0 into buffer 0
    const int m = tid >> 3, kb = (tid & 7) * 4;
    *(v4bf*)&sA[0][m][kb] = *(const v4bf*)&A[(size_t)(row0 + m) * K + kb];
    const int n = tid >> 1, k2 = (tid & 1) * 16;
    const v8bf* src = (const v8bf*)&BT[(size_t)(col0 + n) * K + k2];
    *(v8bf*)&sBt[0][n][k2]     = src[0];
    *(v8bf*)&sBt[0][n][k2 + 8] = src[1];
  }
#endif

#pragma unroll
  for (int it = 0; it < NT; ++it) {
    const int buf = it & 1;
    const int k0  = it * TK;
#if HAVE_TDM
    if (it + 1 < NT) {
      if (wave == 0)
        tdm_load_2d(&BT[(size_t)col0 * K + k0 + TK],
                    (unsigned)(size_t)&sBt[buf ^ 1][0][0], TN, K);
      else if (wave == 1)
        tdm_load_2d(&A[(size_t)row0 * K + k0 + TK],
                    (unsigned)(size_t)&sA[buf ^ 1][0][0], TM, K);
      if (wave < 2) __builtin_amdgcn_s_wait_tensorcnt(1);  // current tile done
    } else {
      if (wave < 2) __builtin_amdgcn_s_wait_tensorcnt(0);  // last tile done
    }
#else
    if (it + 1 < NT) {
      const int m = tid >> 3, kb = (tid & 7) * 4;
      *(v4bf*)&sA[buf ^ 1][m][kb] =
          *(const v4bf*)&A[(size_t)(row0 + m) * K + k0 + TK + kb];
      const int n = tid >> 1, k2 = (tid & 1) * 16;
      const v8bf* src = (const v8bf*)&BT[(size_t)(col0 + n) * K + k0 + TK + k2];
      *(v8bf*)&sBt[buf ^ 1][n][k2]     = src[0];
      *(v8bf*)&sBt[buf ^ 1][n][k2 + 8] = src[1];
    }
#endif
    __syncthreads();   // tile `buf` visible to all waves
    // ---- fragments: 2x b128 each (ISA 7.12.2 layouts) ----
    v8bf a0 = *(const v8bf*)&sA[buf][mn][half * 8];
    v8bf a1 = *(const v8bf*)&sA[buf][mn][16 + half * 8];
    v8bf b0 = *(const v8bf*)&sBt[buf][wave * 16 + mn][half * 8];
    v8bf b1 = *(const v8bf*)&sBt[buf][wave * 16 + mn][16 + half * 8];
    v16bf a = __builtin_shufflevector(a0, a1, 0, 1, 2, 3, 4, 5, 6, 7,
                                      8, 9, 10, 11, 12, 13, 14, 15);
    v16bf b = __builtin_shufflevector(b0, b1, 0, 1, 2, 3, 4, 5, 6, 7,
                                      8, 9, 10, 11, 12, 13, 14, 15);
    acc = __builtin_amdgcn_wmma_f32_16x16x32_bf16(false, a, false, b, (short)0, acc,
                                                  false, false);
    __syncthreads();   // protect `buf` from next-next tile's staging
  }

  if constexpr (!EDOT) {
    const int n  = col0 + wave * 16 + mn;
    const int mb = row0 + half * 8;
#pragma unroll
    for (int r = 0; r < 8; ++r) C[(size_t)(mb + r) * Nd + n] = acc[r];
  } else {
    // head dot + 16-lane tree reduction + 1 atomic per (row, head)
    const int head = blockIdx.y;                    // 64 cols per head, Nd==512
    const float w  = aE[head * 64 + wave * 16 + mn];
    float part[8];
#pragma unroll
    for (int r = 0; r < 8; ++r) part[r] = acc[r] * w;
#pragma unroll
    for (int off = 1; off < 16; off <<= 1) {
#pragma unroll
      for (int r = 0; r < 8; ++r) part[r] += __shfl_xor(part[r], off, 32);
    }
    if (mn == 0) {
      const int mb = row0 + half * 8;
#pragma unroll
      for (int r = 0; r < 8; ++r)
        atomicAdd(&s_e[(size_t)(mb + r) * HHc + head], part[r]);
    }
  }
}

// ---------------- scatter / segment kernels ----------------
__global__ void count_kernel(const int* __restrict__ dst0, float* __restrict__ cnt) {
  int e = blockIdx.x * blockDim.x + threadIdx.x;
  if (e < EEc) atomicAdd(&cnt[dst0[e]], 1.0f);
}

__global__ void scatter_ee_kernel(const int* __restrict__ dst0,
                                  const __bf16* __restrict__ e_all,
                                  float* __restrict__ lacc) {
  int e = blockIdx.x, j = threadIdx.x;  // grid EE, block DE
  int d = dst0[e];
  atomicAdd(&lacc[(size_t)d * DEc + j], (float)e_all[(size_t)e * DEc + j]);
}

__global__ void loop_fill_kernel(const float* __restrict__ lacc,
                                 const float* __restrict__ cnt,
                                 __bf16* __restrict__ e_all) {
  int t = blockIdx.x * blockDim.x + threadIdx.x;
  if (t >= NN * DEc) return;
  int i = t >> 8;
  e_all[(size_t)(EEc + i) * DEc + (t & (DEc - 1))] =
      (__bf16)(lacc[t] / fmaxf(cnt[i], 1.f));
}

__global__ void sdot_kernel(const float* __restrict__ xh,
                            const float* __restrict__ a_src,
                            const float* __restrict__ a_dst,
                            float* __restrict__ s_src, float* __restrict__ s_dst) {
  int t = blockIdx.x * blockDim.x + threadIdx.x;
  if (t >= NN * HHc) return;
  int i = t >> 3, hh = t & 7;
  const float* xr = xh + (size_t)i * DDc + hh * CCc;
  const float* as = a_src + hh * CCc;
  const float* ad = a_dst + hh * CCc;
  float ss = 0.f, sd = 0.f;
#pragma unroll 4
  for (int c = 0; c < CCc; ++c) { float v = xr[c]; ss += v * as[c]; sd += v * ad[c]; }
  s_src[t] = ss; s_dst[t] = sd;
}

__device__ __forceinline__ unsigned f2key(float f) {
  unsigned b = __float_as_uint(f);
  return (b & 0x80000000u) ? ~b : (b | 0x80000000u);
}
__device__ __forceinline__ float key2f(unsigned k) {
  return __uint_as_float((k & 0x80000000u) ? (k & 0x7FFFFFFFu) : ~k);
}

__global__ void logits_kernel(const int* __restrict__ src0, const int* __restrict__ dst0,
                              const float* __restrict__ s_src,
                              const float* __restrict__ s_dst,
                              const float* __restrict__ s_e,
                              float* __restrict__ al_p, unsigned* __restrict__ mx) {
  int t = blockIdx.x * blockDim.x + threadIdx.x;
  if (t >= EALL * HHc) return;
  int e = t >> 3, hh = t & 7;
  int s = e < EEc ? src0[e] : e - EEc;
  int d = e < EEc ? dst0[e] : e - EEc;
  float al = s_src[s * HHc + hh] + s_dst[d * HHc + hh] + s_e[t];
  al = al > 0.f ? al : 0.2f * al;               // leaky_relu(0.2)
  al_p[t] = al;
  atomicMax(&mx[d * HHc + hh], f2key(al));      // segment max via monotone uint map
}

__global__ void softmax_p_kernel(const int* __restrict__ dst0,
                                 const unsigned* __restrict__ mx,
                                 float* __restrict__ al_p, float* __restrict__ ssum) {
  int t = blockIdx.x * blockDim.x + threadIdx.x;
  if (t >= EALL * HHc) return;
  int e = t >> 3, hh = t & 7;
  int d = e < EEc ? dst0[e] : e - EEc;
  float p = expf(al_p[t] - key2f(mx[d * HHc + hh]));
  al_p[t] = p;
  atomicAdd(&ssum[d * HHc + hh], p);
}

__global__ __launch_bounds__(512) void agg_kernel(
    const int* __restrict__ src0, const int* __restrict__ dst0,
    const float* __restrict__ xh, const float* __restrict__ al_p,
    const float* __restrict__ ssum, float* __restrict__ agg) {
  int e = blockIdx.x, j = threadIdx.x;  // grid EALL, block D
  int hh = j >> 6;
  int s = e < EEc ? src0[e] : e - EEc;
  int d = e < EEc ? dst0[e] : e - EEc;
  float alpha = al_p[(size_t)e * HHc + hh] / (ssum[(size_t)d * HHc + hh] + 1e-16f);
  atomicAdd(&agg[(size_t)d * DDc + j], xh[(size_t)s * DDc + j] * alpha);
}

__global__ __launch_bounds__(512) void update_kernel(const float* __restrict__ agg,
                                                     const float* __restrict__ blv,
                                                     float* __restrict__ h) {
  __shared__ float sbuf[16];
  int i = blockIdx.x, j = threadIdx.x;  // grid N, block D
  float v = agg[(size_t)i * DDc + j] + blv[j];
  float mean = block_reduce_sum(v, sbuf) * (1.f / DDc);
  float d = v - mean;
  float var = block_reduce_sum(d * d, sbuf) * (1.f / DDc);
  h[(size_t)i * DDc + j] += d * rsqrtf(var + 1e-5f);
}

__global__ __launch_bounds__(512) void pool_kernel(const float* __restrict__ h,
                                                   const int* __restrict__ batch,
                                                   float* __restrict__ h_pool) {
  int i = blockIdx.x, j = threadIdx.x;  // grid N, block D
  atomicAdd(&h_pool[(size_t)batch[i] * DDc + j], h[(size_t)i * DDc + j]);
}

__global__ void gf_kernel(const float* __restrict__ nA, const float* __restrict__ nB,
                          const float* __restrict__ ssz, float* __restrict__ gf) {
  int g = threadIdx.x;
  if (g >= GGc) return;
  float inv = 1.f / (ssz[g] + 1e-10f);
  gf[2 * g]     = nA[g] * inv;
  gf[2 * g + 1] = nB[g] * inv;
}

__global__ void concat_kernel(const float* __restrict__ h_r, const float* __restrict__ gm,
                              float* __restrict__ comb) {
  int t = blockIdx.x * blockDim.x + threadIdx.x;
  if (t >= GGc * (DDc + DEc)) return;
  int g = t / (DDc + DEc), j = t - g * (DDc + DEc);
  comb[t] = (j < DDc) ? h_r[(size_t)g * DDc + j] : gm[(size_t)g * DEc + (j - DDc)];
}

__global__ void final_dot_kernel(const float* __restrict__ f1,
                                 const float* __restrict__ Wf2,
                                 const float* __restrict__ bf2, float* __restrict__ out) {
  __shared__ float sbuf[16];
  int g = blockIdx.x, j = threadIdx.x;  // grid G, block DE
  float v = f1[(size_t)g * DEc + j] * Wf2[j];
  float s = block_reduce_sum(v, sbuf);
  if (j == 0) out[g] = s + bf2[0];
}

// ---------------- launcher ----------------
extern "C" void kernel_launch(void* const* d_in, const int* in_sizes, int n_in,
                              void* d_out, int out_size, void* d_ws, size_t ws_size,
                              hipStream_t stream) {
  (void)in_sizes; (void)n_in; (void)out_size; (void)ws_size;
  const float* x    = (const float*)d_in[0];
  const float* ea   = (const float*)d_in[1];
  const float* nA   = (const float*)d_in[2];
  const float* nB   = (const float*)d_in[3];
  const float* ssz  = (const float*)d_in[4];
  const float* Wn   = (const float*)d_in[5];
  const float* bn   = (const float*)d_in[6];
  const float* Wee  = (const float*)d_in[7];
  const float* bee  = (const float*)d_in[8];
  const float* Wl   = (const float*)d_in[9];
  const float* a_src= (const float*)d_in[10];
  const float* a_dst= (const float*)d_in[11];
  const float* Wle  = (const float*)d_in[12];
  const float* a_e  = (const float*)d_in[13];
  const float* bl   = (const float*)d_in[14];
  const float* Wp   = (const float*)d_in[15];
  const float* bp   = (const float*)d_in[16];
  const float* Wg1  = (const float*)d_in[17];
  const float* bg1  = (const float*)d_in[18];
  const float* Wg2  = (const float*)d_in[19];
  const float* bg2  = (const float*)d_in[20];
  const float* Wf1  = (const float*)d_in[21];
  const float* bf1  = (const float*)d_in[22];
  const float* Wf2  = (const float*)d_in[23];
  const float* bf2  = (const float*)d_in[24];
  const int*   ei   = (const int*)d_in[25];
  const int*   batch= (const int*)d_in[26];
  const int* src0 = ei;
  const int* dst0 = ei + EEc;

  // ---- workspace carve-up (~188 MB) ----
  char* wsb = (char*)d_ws;
  size_t off = 0;
  auto allocf = [&](size_t n) { float* p = (float*)(wsb + off); off += n * sizeof(float); return p; };
  auto allocb = [&](size_t n) {   // bf16, keep 64B alignment for vector/TDM loads
    off = (off + 63) & ~(size_t)63;
    __bf16* p = (__bf16*)(wsb + off); off += n * sizeof(__bf16); return p; };
  float*    h      = allocf((size_t)NN * DDc);
  float*    xh     = allocf((size_t)NN * DDc);
  float*    agg    = allocf((size_t)NN * DDc);
  float*    lacc   = allocf((size_t)NN * DEc);
  float*    cnt    = allocf(NN);
  float*    s_src  = allocf((size_t)NN * HHc);
  float*    s_dst  = allocf((size_t)NN * HHc);
  float*    s_e    = allocf((size_t)EALL * HHc);
  float*    al_p   = allocf((size_t)EALL * HHc);
  float*    ssum   = allocf((size_t)NN * HHc);
  unsigned* mx     = (unsigned*)allocf((size_t)NN * HHc);
  float*    h_pool = allocf((size_t)GGc * DDc);
  float*    h_r    = allocf((size_t)GGc * DDc);
  float*    gf     = allocf((size_t)GGc * 2);
  float*    gm1    = allocf((size_t)GGc * DEc);
  float*    gm     = allocf((size_t)GGc * DEc);
  float*    comb   = allocf((size_t)GGc * (DDc + DEc));
  float*    f1     = allocf((size_t)GGc * DEc);
  __bf16*   e_all  = allocb((size_t)EALL * DEc);          // edge features, bf16
  __bf16*   h_bf   = allocb((size_t)NN * DDc);            // h in bf16 (per layer)
  __bf16*   WlT    = allocb((size_t)LLc * DDc * DDc);     // Wl^T  [l][n=512][k=512]
  __bf16*   WleT   = allocb((size_t)LLc * DDc * DEc);     // Wle^T [l][n=512][k=256]

  // ---- weight pre-transpose + bf16 conversion (runs every call; deterministic) ----
  for (int l = 0; l < LLc; ++l) {
    transpose_cvt_kernel<<<CDIV(DDc * DDc, 256), 256, 0, stream>>>(
        Wl + (size_t)l * DDc * DDc, WlT + (size_t)l * DDc * DDc, DDc, DDc);
    transpose_cvt_kernel<<<CDIV(DEc * DDc, 256), 256, 0, stream>>>(
        Wle + (size_t)l * DEc * DDc, WleT + (size_t)l * DDc * DEc, DEc, DDc);
  }

  // ---- embeddings ----
  hipMemsetAsync(lacc, 0, (size_t)NN * DEc * 4, stream);
  hipMemsetAsync(cnt, 0, NN * 4, stream);
  hipMemsetAsync(h_pool, 0, (size_t)GGc * DDc * 4, stream);

  linear_ln_elu_kernel<float><<<NN, DDc, 0, stream>>>(x, 4, 0, 4, Wn, bn, h);
  linear_ln_elu_kernel<__bf16><<<EEc, DEc, 0, stream>>>(ea, 3, 1, 2, Wee, bee, e_all);
  count_kernel<<<CDIV(EEc, 256), 256, 0, stream>>>(dst0, cnt);
  scatter_ee_kernel<<<EEc, DEc, 0, stream>>>(dst0, e_all, lacc);
  loop_fill_kernel<<<CDIV(NN * DEc, 256), 256, 0, stream>>>(lacc, cnt, e_all);

  // ---- GAT layers ----
  for (int l = 0; l < LLc; ++l) {
    const __bf16* WlT_l  = WlT + (size_t)l * DDc * DDc;
    const __bf16* WleT_l = WleT + (size_t)l * DDc * DEc;
    const float* as_l = a_src + (size_t)l * HHc * CCc;
    const float* ad_l = a_dst + (size_t)l * HHc * CCc;
    const float* ae_l = a_e + (size_t)l * HHc * CCc;
    const float* bl_l = bl + (size_t)l * DDc;

    cvt_bf16_kernel<<<CDIV(NN * DDc / 4, 256), 256, 0, stream>>>(h, h_bf, NN * DDc / 4);
    gemm_wmma_kernel<DDc, false><<<dim3(NN / TM, DDc / TN), 128, 0, stream>>>(
        h_bf, WlT_l, xh, nullptr, nullptr, DDc);
    sdot_kernel<<<CDIV(NN * HHc, 256), 256, 0, stream>>>(xh, as_l, ad_l, s_src, s_dst);

    hipMemsetAsync(s_e, 0, (size_t)EALL * HHc * 4, stream);
    gemm_wmma_kernel<DEc, true><<<dim3(EALL / TM, DDc / TN), 128, 0, stream>>>(
        e_all, WleT_l, nullptr, ae_l, s_e, DDc);

    hipMemsetAsync(mx, 0, (size_t)NN * HHc * 4, stream);   // 0 == key floor
    hipMemsetAsync(ssum, 0, (size_t)NN * HHc * 4, stream);
    hipMemsetAsync(agg, 0, (size_t)NN * DDc * 4, stream);

    logits_kernel<<<CDIV(EALL * HHc, 256), 256, 0, stream>>>(src0, dst0, s_src, s_dst,
                                                             s_e, al_p, mx);
    softmax_p_kernel<<<CDIV(EALL * HHc, 256), 256, 0, stream>>>(dst0, mx, al_p, ssum);
    agg_kernel<<<EALL, DDc, 0, stream>>>(src0, dst0, xh, al_p, ssum, agg);
    update_kernel<<<NN, DDc, 0, stream>>>(agg, bl_l, h);
  }

  // ---- readout ----
  pool_kernel<<<NN, DDc, 0, stream>>>(h, batch, h_pool);
  linear_ln_elu_kernel<float><<<GGc, DDc, 0, stream>>>(h_pool, DDc, 0, DDc, Wp, bp, h_r);
  gf_kernel<<<1, GGc, 0, stream>>>(nA, nB, ssz, gf);
  linear_ln_elu_kernel<float><<<GGc, DEc, 0, stream>>>(gf, 2, 0, 2, Wg1, bg1, gm1);
  linear_ln_elu_kernel<float><<<GGc, DEc, 0, stream>>>(gm1, DEc, 0, DEc, Wg2, bg2, gm);
  concat_kernel<<<CDIV(GGc * (DDc + DEc), 256), 256, 0, stream>>>(h_r, gm, comb);
  linear_ln_elu_kernel<float><<<GGc, DEc, 0, stream>>>(comb, DDc + DEc, 0, DDc + DEc,
                                                       Wf1, bf1, f1);
  final_dot_kernel<<<GGc, DEc, 0, stream>>>(f1, Wf2, bf2, (float*)d_out);
}
